// GCN_53635551592510
// MI455X (gfx1250) — matrix operand order repaired and checked by reading, not compiled, not run on hardware
//
#include <hip/hip_runtime.h>
#include <hip/hip_bf16.h>
#include <math.h>

// ---------------------------------------------------------------------------
// Types for CDNA5 WMMA (wave32): 16x16x32 bf16 -> f32
// ---------------------------------------------------------------------------
typedef __attribute__((ext_vector_type(16))) __bf16 v16bf;
typedef __attribute__((ext_vector_type(8)))  float  v8f;
typedef __attribute__((ext_vector_type(2)))  __bf16 bf16x2;

#define D 256
#define KTILES 8   // 256 / 32

// ---- A fragment: CDNA5 ISA "16-bit A-Matrix 16x32" table (interleaved K) ----
//   lanes 0-15 : V0..V3 -> K=0..7,  V4..V7 -> K=16..23
//   lanes 16-31: V0..V3 -> K=8..15, V4..V7 -> K=24..31
__device__ __forceinline__ int fragA_kbase(int v, int hseg) {
    return (v < 4) ? (hseg * 8 + 2 * v) : (16 + hseg * 8 + 2 * (v - 4));
}

__device__ __forceinline__ v16bf load_fragA(const __bf16* base, int hseg) {
    v16bf f;
#pragma unroll
    for (int v = 0; v < 8; ++v) {
        const int kb = fragA_kbase(v, hseg);
        bf16x2 p = *reinterpret_cast<const bf16x2*>(base + kb);
        f[2 * v]     = p.x;
        f[2 * v + 1] = p.y;
    }
    return f;
}

// ---- B fragment: 32x16 B, same interleaved K order as the A fragment ----
//   lane holds column n = lane&15; lanes 0-15 hold K=0..7,16..23, lanes 16-31 K=8..15,24..31.
//   With B stored transposed as Bt[N][K], each lane reads pairs at
//   (row n)*D + k0 + fragA_kbase(v, hseg).
__device__ __forceinline__ v16bf load_fragB(const __bf16* base, int hseg) {
    const __bf16* p = base;
    v16bf f;
#pragma unroll
    for (int v = 0; v < 8; ++v) {
        bf16x2 q = *reinterpret_cast<const bf16x2*>(p + fragA_kbase(v, hseg));
        f[2 * v]     = q.x;
        f[2 * v + 1] = q.y;
    }
    return f;
}

// On-the-fly A fragment for the predictor: bf16(h2[e0,k] * h2[e1,k])
__device__ __forceinline__ v16bf build_fragA_mul(const float* __restrict__ A0,
                                                 const float* __restrict__ A1,
                                                 int k0, int hseg) {
    v16bf a;
#pragma unroll
    for (int v = 0; v < 8; ++v) {
        const int kb = fragA_kbase(v, hseg);
        float2 p0 = *reinterpret_cast<const float2*>(A0 + k0 + kb);
        float2 p1 = *reinterpret_cast<const float2*>(A1 + k0 + kb);
        a[2 * v]     = (__bf16)(p0.x * p1.x);
        a[2 * v + 1] = (__bf16)(p0.y * p1.y);
    }
    return a;
}

__device__ __forceinline__ void atom_add_f32(float* p, float v) {
#if defined(__HIP_DEVICE_COMPILE__)
    unsafeAtomicAdd(p, v);   // native global_atomic_add_f32 (unordered fp add)
#else
    atomicAdd(p, v);
#endif
}

__device__ __forceinline__ v8f wmma_bf16(v16bf a, v16bf b, v8f c) {
    return __builtin_amdgcn_wmma_f32_16x16x32_bf16(false, a, false, b,
                                                   (short)0, c, false, false);
}

// ---------------------------------------------------------------------------
// Elementwise / setup kernels
// ---------------------------------------------------------------------------
__global__ void k_zero(float* p, long n) {
    long i = (long)blockIdx.x * blockDim.x + threadIdx.x;
    if (i < n) p[i] = 0.0f;
}

__global__ void k_deg(const int* __restrict__ row, float* deg, int E) {
    int e = blockIdx.x * blockDim.x + threadIdx.x;
    if (e < E) atom_add_f32(&deg[row[e]], 1.0f);
}

__global__ void k_dinv(const float* __restrict__ deg, float* dinv, int N) {
    int i = blockIdx.x * blockDim.x + threadIdx.x;
    if (i < N) dinv[i] = rsqrtf(deg[i] + 1.0f);
}

// h = concat(emb[N,128], x[N,128]) converted to bf16
__global__ void k_build_h(const float* __restrict__ emb, const float* __restrict__ x,
                          __bf16* __restrict__ h, int N) {
    long idx = (long)blockIdx.x * blockDim.x + threadIdx.x;
    if (idx >= (long)N * D) return;
    int i = (int)(idx >> 8);
    int j = (int)(idx & 255);
    float v = (j < 128) ? emb[(size_t)i * 128 + j] : x[(size_t)i * 128 + (j - 128)];
    h[idx] = (__bf16)v;
}

// Wt[n*256 + k] = bf16(W[k*256 + n])   (transpose so B-fragment loads are contiguous)
__global__ void k_convert_wt(const float* __restrict__ W, __bf16* __restrict__ Wt) {
    int idx = blockIdx.x * blockDim.x + threadIdx.x;   // 65536 threads
    int n = idx >> 8;
    int k = idx & 255;
    Wt[(size_t)n * D + k] = (__bf16)W[(size_t)k * D + n];
}

// ---------------------------------------------------------------------------
// Dense GEMM: C[M,256] = A_bf[M,256] @ B (B given as Bt[N=256][K=256] bf16)
// One wave computes a 16x64 strip (4 accumulators). Software-pipelined:
// the NEXT K-step's A + 4 B fragments are loaded before the CURRENT step's
// WMMAs; __builtin_amdgcn_sched_barrier(0) pins that order so the machine
// scheduler cannot sink the loads back to their consumers, keeping ~10 VMEM
// loads in flight behind the matrix pipe (partial s_wait_loadcnt).
// ---------------------------------------------------------------------------
__global__ __launch_bounds__(32)
void k_gemm_wmma(const __bf16* __restrict__ A, const __bf16* __restrict__ Bt,
                 float* __restrict__ C, int M) {
    const int mt   = blockIdx.x;
    const int ng   = blockIdx.y;        // 0..3, covers N-tiles ng*4 .. ng*4+3
    const int lane = threadIdx.x & 31;
    const int r    = lane & 15;
    const int hseg = lane >> 4;

    int arow = mt * 16 + r;
    if (arow >= M) arow = M - 1;
    const __bf16* ap = A + (size_t)arow * D;
    const __bf16* bp0 = Bt + (size_t)((ng * 4 + 0) * 16 + r) * D;
    const __bf16* bp1 = Bt + (size_t)((ng * 4 + 1) * 16 + r) * D;
    const __bf16* bp2 = Bt + (size_t)((ng * 4 + 2) * 16 + r) * D;
    const __bf16* bp3 = Bt + (size_t)((ng * 4 + 3) * 16 + r) * D;

    v8f zero = {};
    v8f acc0 = zero, acc1 = zero, acc2 = zero, acc3 = zero;

    // prologue: fragments for K-step 0
    v16bf a  = load_fragA(ap, hseg);
    v16bf b0 = load_fragB(bp0, hseg);
    v16bf b1 = load_fragB(bp1, hseg);
    v16bf b2 = load_fragB(bp2, hseg);
    v16bf b3 = load_fragB(bp3, hseg);

#pragma unroll
    for (int kt = 0; kt < KTILES; ++kt) {
        v16bf an = {}, n0 = {}, n1 = {}, n2 = {}, n3 = {};
        if (kt < KTILES - 1) {
            const int kn = (kt + 1) * 32;
            an = load_fragA(ap + kn, hseg);
            n0 = load_fragB(bp0 + kn, hseg);
            n1 = load_fragB(bp1 + kn, hseg);
            n2 = load_fragB(bp2 + kn, hseg);
            n3 = load_fragB(bp3 + kn, hseg);
        }
        // Pin schedule: next-step loads stay ABOVE, current WMMAs BELOW.
        __builtin_amdgcn_sched_barrier(0);
        acc0 = wmma_bf16(a, b0, acc0);
        acc1 = wmma_bf16(a, b1, acc1);
        acc2 = wmma_bf16(a, b2, acc2);
        acc3 = wmma_bf16(a, b3, acc3);
        __builtin_amdgcn_sched_barrier(0);
        if (kt < KTILES - 1) { a = an; b0 = n0; b1 = n1; b2 = n2; b3 = n3; }
    }

    // C/D layout: VGPR i -> row = i + 8*hseg, col = lane&15
    v8f accs[4] = {acc0, acc1, acc2, acc3};
#pragma unroll
    for (int t = 0; t < 4; ++t) {
        const int ncol = (ng * 4 + t) * 16 + r;
#pragma unroll
        for (int i = 0; i < 8; ++i) {
            const int rr = mt * 16 + i + 8 * hseg;
            if (rr < M) C[(size_t)rr * D + ncol] = accs[t][i];
        }
    }
}

// ---------------------------------------------------------------------------
// GCN aggregation:  agg = dinv^2 * m + b   then edge scatter-add
// ---------------------------------------------------------------------------
__global__ void k_init_agg(const float* __restrict__ m, const float* __restrict__ dinv,
                           const float* __restrict__ bias, float* __restrict__ agg, int N) {
    long idx = (long)blockIdx.x * blockDim.x + threadIdx.x;
    if (idx >= (long)N * D) return;
    int i = (int)(idx >> 8);
    int j = (int)(idx & 255);
    float di = dinv[i];
    agg[idx] = di * di * m[idx] + bias[j];
}

__global__ void k_scatter(const float* __restrict__ m, const int* __restrict__ erow,
                          const int* __restrict__ ecol, const float* __restrict__ dinv,
                          float* __restrict__ agg, long E) {
    long idx = (long)blockIdx.x * blockDim.x + threadIdx.x;
    if (idx >= E * 64) return;
    int e  = (int)(idx >> 6);
    int cg = (int)(idx & 63) * 4;
    int r = erow[e], c = ecol[e];
    float w = dinv[r] * dinv[c];
    float4 v = *reinterpret_cast<const float4*>(m + (size_t)c * D + cg);
    float* dst = agg + (size_t)r * D + cg;
    atom_add_f32(dst + 0, w * v.x);
    atom_add_f32(dst + 1, w * v.y);
    atom_add_f32(dst + 2, w * v.z);
    atom_add_f32(dst + 3, w * v.w);
}

// ---------------------------------------------------------------------------
// BatchNorm: one block per column; column mean / rsqrt(var+eps)
// ---------------------------------------------------------------------------
__global__ __launch_bounds__(256)
void k_bn_stats(const float* __restrict__ agg, int N,
                float* __restrict__ mean, float* __restrict__ rinv) {
    __shared__ float ss[256];
    __shared__ float sq[256];
    const int j = blockIdx.x;
    const int t = threadIdx.x;
    float s = 0.0f, q = 0.0f;
    for (int i = t; i < N; i += 256) {
        float v = agg[(size_t)i * D + j];
        s += v; q += v * v;
    }
    ss[t] = s; sq[t] = q;
    __syncthreads();
    for (int o = 128; o > 0; o >>= 1) {
        if (t < o) { ss[t] += ss[t + o]; sq[t] += sq[t + o]; }
        __syncthreads();
    }
    if (t == 0) {
        float mu  = ss[0] / (float)N;
        float var = sq[0] / (float)N - mu * mu;
        mean[j] = mu;
        rinv[j] = rsqrtf(var + 1e-5f);
    }
}

__global__ void k_bn_apply(const float* __restrict__ agg, const float* __restrict__ mean,
                           const float* __restrict__ rinv, const float* __restrict__ gamma,
                           const float* __restrict__ beta, __bf16* __restrict__ h, int N) {
    long idx = (long)blockIdx.x * blockDim.x + threadIdx.x;
    if (idx >= (long)N * D) return;
    int j = (int)(idx & 255);
    float v = (agg[idx] - mean[j]) * rinv[j] * gamma[j] + beta[j];
    h[idx] = (__bf16)(v > 0.0f ? v : 0.0f);
}

// ---------------------------------------------------------------------------
// Fused link predictor: out[q] = sigmoid( relu( (h2[e0]*h2[e1]) @ pW1 + pb1 ) @ pW2 + pb2 )
// One wave per 16-row tile; A fragment built on the fly as bf16(h2[e0,k]*h2[e1,k]);
// 16 N-tile accumulators per wave. B-fragment groups are double-buffered
// (group g+1 prefetched before group g's WMMAs, rolling into the next K-step),
// the next K-step's A fragment is prefetched mid-stream, and sched_barrier(0)
// pins prefetch-above / WMMA-below so loads stay in flight.
// Epilogue: bias+ReLU+dot(pW2), cross-lane half-reduction, sigmoid, store.
// ---------------------------------------------------------------------------
__global__ __launch_bounds__(32)
void k_predict_wmma(const float* __restrict__ h2, const int* __restrict__ edges, int Q,
                    const __bf16* __restrict__ pW1t, const float* __restrict__ pb1,
                    const float* __restrict__ pW2, const float* __restrict__ pb2,
                    float* __restrict__ out) {
    const int qt   = blockIdx.x;
    const int lane = threadIdx.x & 31;
    const int r    = lane & 15;
    const int hseg = lane >> 4;

    int q = qt * 16 + r;
    if (q >= Q) q = Q - 1;
    const float* A0 = h2 + (size_t)edges[q]     * D;
    const float* A1 = h2 + (size_t)edges[Q + q] * D;

    v8f zero = {};
    v8f acc[16];
#pragma unroll
    for (int t = 0; t < 16; ++t) acc[t] = zero;

    // B row base for this lane's column within each N-tile group
    const __bf16* bbase = pW1t + (size_t)r * D;

    // prologue: A for K-step 0, B group 0 at K-step 0
    v16bf a = build_fragA_mul(A0, A1, 0, hseg);
    v16bf b0 = load_fragB(bbase + (size_t)(0 * 16) * D, hseg);
    v16bf b1 = load_fragB(bbase + (size_t)(1 * 16) * D, hseg);
    v16bf b2 = load_fragB(bbase + (size_t)(2 * 16) * D, hseg);
    v16bf b3 = load_fragB(bbase + (size_t)(3 * 16) * D, hseg);

#pragma unroll
    for (int kt = 0; kt < KTILES; ++kt) {
        const int k0 = kt * 32;
        v16bf a_nxt = {};
#pragma unroll
        for (int g = 0; g < 4; ++g) {
            // prefetch next B group (rolls into next K-step at g==3)
            const bool have_next = !(kt == KTILES - 1 && g == 3);
            const int gg = (g == 3) ? 0 : (g + 1);
            const int gk = (g == 3) ? (k0 + 32) : k0;
            v16bf n0 = {}, n1 = {}, n2 = {}, n3 = {};
            if (have_next) {
                const __bf16* nb = bbase + (size_t)(gg * 4) * 16 * D + gk;
                n0 = load_fragB(nb + (size_t)(0 * 16) * D, hseg);
                n1 = load_fragB(nb + (size_t)(1 * 16) * D, hseg);
                n2 = load_fragB(nb + (size_t)(2 * 16) * D, hseg);
                n3 = load_fragB(nb + (size_t)(3 * 16) * D, hseg);
            }
            if (g == 1 && kt < KTILES - 1)
                a_nxt = build_fragA_mul(A0, A1, k0 + 32, hseg);

            // Pin schedule: prefetches stay ABOVE, this group's WMMAs BELOW.
            __builtin_amdgcn_sched_barrier(0);
            const int nt = g * 4;
            acc[nt + 0] = wmma_bf16(a, b0, acc[nt + 0]);
            acc[nt + 1] = wmma_bf16(a, b1, acc[nt + 1]);
            acc[nt + 2] = wmma_bf16(a, b2, acc[nt + 2]);
            acc[nt + 3] = wmma_bf16(a, b3, acc[nt + 3]);
            __builtin_amdgcn_sched_barrier(0);
            if (have_next) { b0 = n0; b1 = n1; b2 = n2; b3 = n3; }
        }
        if (kt < KTILES - 1) a = a_nxt;
    }

    // bias + ReLU + multiply by pW2, accumulate per-lane row partials
    float part[8];
#pragma unroll
    for (int i = 0; i < 8; ++i) part[i] = 0.0f;
#pragma unroll
    for (int nt = 0; nt < 16; ++nt) {
        const int col = nt * 16 + r;
        const float bias = pb1[col];
        const float w2   = pW2[col];
#pragma unroll
        for (int i = 0; i < 8; ++i) {
            float v = acc[nt][i] + bias;
            v = v > 0.0f ? v : 0.0f;
            part[i] += v * w2;
        }
    }
    // reduce across the 16 lanes of each half (masks 1,2,4,8 stay within half)
#pragma unroll
    for (int m = 1; m < 16; m <<= 1) {
#pragma unroll
        for (int i = 0; i < 8; ++i) part[i] += __shfl_xor(part[i], m, 32);
    }
    if (r == 0) {
        const float b2 = pb2[0];
#pragma unroll
        for (int i = 0; i < 8; ++i) {
            const int qq = qt * 16 + i + 8 * hseg;
            if (qq < Q) out[qq] = 1.0f / (1.0f + __expf(-(part[i] + b2)));
        }
    }
}

// ---------------------------------------------------------------------------
// Host launcher
// ---------------------------------------------------------------------------
extern "C" void kernel_launch(void* const* d_in, const int* in_sizes, int n_in,
                              void* d_out, int out_size, void* d_ws, size_t ws_size,
                              hipStream_t stream) {
    const float* x      = (const float*)d_in[0];
    const int*   adj_r  = (const int*)  d_in[1];
    const int*   adj_c  = (const int*)  d_in[2];
    const int*   edges  = (const int*)  d_in[3];
    const float* emb    = (const float*)d_in[4];
    const float* W1     = (const float*)d_in[5];
    const float* b1     = (const float*)d_in[6];
    const float* W2     = (const float*)d_in[7];
    const float* b2     = (const float*)d_in[8];
    const float* gamma  = (const float*)d_in[9];
    const float* beta   = (const float*)d_in[10];
    const float* pW1    = (const float*)d_in[11];
    const float* pb1    = (const float*)d_in[12];
    const float* pW2    = (const float*)d_in[13];
    const float* pb2    = (const float*)d_in[14];
    float*       out    = (float*)d_out;

    const int  N = in_sizes[0] / 128;
    const long E = in_sizes[1];
    const int  Q = in_sizes[3] / 2;

    // ---- workspace carve-up (256B aligned) ----
    char* w = (char*)d_ws;
    size_t off = 0;
    auto alloc = [&](size_t bytes) -> char* {
        char* p = w + off;
        off += (bytes + 255) & ~(size_t)255;
        return p;
    };
    float*  deg   = (float*) alloc((size_t)N * 4);
    float*  dinv  = (float*) alloc((size_t)N * 4);
    __bf16* h_bf  = (__bf16*)alloc((size_t)N * D * 2);   // h0 then h1 (reused)
    float*  m     = (float*) alloc((size_t)N * D * 4);   // m1 then m2 (reused)
    float*  agg   = (float*) alloc((size_t)N * D * 4);   // agg1 then h2 (reused)
    __bf16* W1t   = (__bf16*)alloc((size_t)D * D * 2);
    __bf16* W2t   = (__bf16*)alloc((size_t)D * D * 2);
    __bf16* pW1t  = (__bf16*)alloc((size_t)D * D * 2);
    float*  mean  = (float*) alloc(D * 4);
    float*  rinv  = (float*) alloc(D * 4);
    (void)ws_size; (void)n_in; (void)out_size;

    const long ND   = (long)N * D;
    const int  bND  = (int)((ND + 255) / 256);
    const int  bN   = (N + 255) / 256;
    const int  bE   = (int)((E + 255) / 256);
    const long Esc  = E * 64;
    const int  bEsc = (int)((Esc + 255) / 256);
    const int  Mt   = (N + 15) / 16;
    const int  Qt   = (Q + 15) / 16;

    // degrees + dinv
    k_zero<<<bN, 256, 0, stream>>>(deg, N);
    k_deg<<<bE, 256, 0, stream>>>(adj_r, deg, (int)E);
    k_dinv<<<bN, 256, 0, stream>>>(deg, dinv, N);

    // h0 = concat(emb, x) -> bf16 ; weight transposes -> bf16
    k_build_h<<<bND, 256, 0, stream>>>(emb, x, h_bf, N);
    k_convert_wt<<<(D * D) / 256, 256, 0, stream>>>(W1, W1t);
    k_convert_wt<<<(D * D) / 256, 256, 0, stream>>>(W2, W2t);
    k_convert_wt<<<(D * D) / 256, 256, 0, stream>>>(pW1, pW1t);

    // ---- conv1 ----
    k_gemm_wmma<<<dim3(Mt, 4), 32, 0, stream>>>(h_bf, W1t, m, N);
    k_init_agg<<<bND, 256, 0, stream>>>(m, dinv, b1, agg, N);
    k_scatter<<<bEsc, 256, 0, stream>>>(m, adj_r, adj_c, dinv, agg, E);

    // ---- BN + ReLU -> h1 (bf16, reuses h_bf) ----
    k_bn_stats<<<D, 256, 0, stream>>>(agg, N, mean, rinv);
    k_bn_apply<<<bND, 256, 0, stream>>>(agg, mean, rinv, gamma, beta, h_bf, N);

    // ---- conv2 (reuse m and agg buffers) ----
    k_gemm_wmma<<<dim3(Mt, 4), 32, 0, stream>>>(h_bf, W2t, m, N);
    k_init_agg<<<bND, 256, 0, stream>>>(m, dinv, b2, agg, N);
    k_scatter<<<bEsc, 256, 0, stream>>>(m, adj_r, adj_c, dinv, agg, E);

    // ---- fused link predictor (agg now holds h2) ----
    k_predict_wmma<<<Qt, 32, 0, stream>>>(agg, edges, Q, pW1t, pb1, pW2, pb2, out);
}
